// RFDM_53223234732094
// MI455X (gfx1250) — matrix operand order repaired and checked by reading, not compiled, not run on hardware
//
#include <hip/hip_runtime.h>
#include <hip/hip_bf16.h>
#include <math.h>

#define DEV __device__ __forceinline__

typedef __attribute__((ext_vector_type(16))) _Float16 v16h;
typedef __attribute__((ext_vector_type(8)))  float    v8f;
typedef unsigned int u32x4 __attribute__((ext_vector_type(4)));
typedef int          i32x4 __attribute__((ext_vector_type(4)));
typedef int          i32x8 __attribute__((ext_vector_type(8)));

constexpr int Bn = 16, Cc = 64, Hh = 128, Ww = 128, HID = 128, DC = 16;
constexpr int HW = Hh * Ww;          // 16384
constexpr int RG = 18, RR = RG * RG; // 18x18 halo region = 324 positions

#if __has_builtin(__builtin_amdgcn_tensor_load_to_lds) && \
    __has_builtin(__builtin_amdgcn_s_wait_tensorcnt)
#define HAVE_TDM 1
#else
#define HAVE_TDM 0
#endif

// ---- WMMA helpers (CDNA5 wave32, 16x16x32 f16 -> f32) ----------------------

DEV v8f wmma16(v16h a, v16h b, v8f c) {
  // (neg_a, A, neg_b, B, c_mod, C, reuse_a, reuse_b)
  return __builtin_amdgcn_wmma_f32_16x16x32_f16(false, a, false, b, (short)0, c,
                                                false, false);
}

// A operand: 16x32 (MxK), W stored row-major [M][ks] in LDS (f16).
// lane<16 holds M=lane, K in {0..15}; lane>=16 holds M=lane-16, K in {16..31}.
DEV v16h ldsA(const _Float16* Wt, int ks, int mBase, int kBase, int lane) {
  int m = mBase + (lane & 15);
  int kh = (lane < 16) ? 0 : 16;
  v16h a;
#pragma unroll
  for (int e = 0; e < 16; ++e) {
    int k = kBase + kh + e;
    a[e] = Wt[m * ks + k];
  }
  return a;
}

// B operand: 32x16 (KxN) from channel-strided f16 source: B(k,n) = base[k*stride+n].
DEV v16h ldB(const _Float16* base, int kBase, int stride, int lane) {
  int n = lane & 15;
  int k0 = kBase + ((lane < 16) ? 0 : 16);
  v16h r;
#pragma unroll
  for (int e = 0; e < 16; ++e) r[e] = base[(long)(k0 + e) * stride + n];
  return r;
}

// B operand from TRANSPOSED source laid out [pos][ch] (row stride = `stride`
// halves): each lane reads 16 contiguous halves -> 2x ds_load_b128.
DEV v16h ldBt(const _Float16* base, int kBase, int stride, int lane) {
  int n = lane & 15;
  int k0 = kBase + ((lane < 16) ? 0 : 16);
  const _Float16* p = base + (long)n * stride + k0;
  v16h r;
#pragma unroll
  for (int e = 0; e < 16; ++e) r[e] = p[e];
  return r;
}

DEV float fast_tanh(float x) {
#if __has_builtin(__builtin_amdgcn_tanhf)
  return __builtin_amdgcn_tanhf(x);        // v_tanh_f32 (CDNA5 TRANS op)
#elif __has_builtin(__builtin_amdgcn_tanh_f32)
  return __builtin_amdgcn_tanh_f32(x);
#else
  return tanhf(x);
#endif
}

DEV float sigmoidf(float x) { return 1.0f / (1.0f + __expf(-x)); }
DEV float geluf(float x) { return 0.5f * x * (1.0f + erff(x * 0.70710678118f)); }

// ---- fp32 -> f16 ingest ----------------------------------------------------

__global__ void k_cvt(const float* __restrict__ x, _Float16* __restrict__ y, long n) {
  long i = (long)blockIdx.x * blockDim.x + threadIdx.x;
  long st = (long)gridDim.x * blockDim.x;
  for (; i < n; i += st) y[i] = (_Float16)x[i];
}

// ---- Gate: x = (1-sigmoid(Wf2 tanh(Wf1 x+b)+b)) * tanh(Wz2 tanh(Wz1 x+b)+b)
// Pure pointwise: 4 chained 64x64 WMMA matmuls over 16-position tiles. -------

__global__ void __launch_bounds__(256) k_gate(
    const _Float16* __restrict__ xin,
    const float* __restrict__ wz1, const float* __restrict__ bz1,
    const float* __restrict__ wz2, const float* __restrict__ bz2,
    const float* __restrict__ wf1, const float* __restrict__ bf1,
    const float* __restrict__ wf2, const float* __restrict__ bf2,
    _Float16* __restrict__ xg) {
  __shared__ _Float16 smW[4][64 * 64];  // f16 weights, A-layout [o][c]
  __shared__ float    smB[4][64];
  __shared__ _Float16 smT[8][16 * 64];  // per-wave intermediate, TRANSPOSED [pos][ch]
  int tid = threadIdx.x;
  for (int i = tid; i < 4096; i += 256) {
    smW[0][i] = (_Float16)wz1[i];
    smW[1][i] = (_Float16)wz2[i];
    smW[2][i] = (_Float16)wf1[i];
    smW[3][i] = (_Float16)wf2[i];
  }
  if (tid < 64) {
    smB[0][tid] = bz1[tid]; smB[1][tid] = bz2[tid];
    smB[2][tid] = bf1[tid]; smB[3][tid] = bf2[tid];
  }
  __syncthreads();

  int lane = tid & 31, wv = tid >> 5;
  long pbase = ((long)blockIdx.x * 8 + wv) * 16;  // flat position base
  int b  = (int)(pbase >> 14);
  int pi = (int)(pbase & (HW - 1));
  const _Float16* xp = xin + (long)b * Cc * HW + pi;
  __builtin_prefetch(xp, 0, 1);  // global_prefetch_b8
  v16h xb0 = ldB(xp, 0, HW, lane);
  v16h xb1 = ldB(xp, 32, HW, lane);
  _Float16* t1 = smT[wv];
  int n = lane & 15;
  int rsel = (lane < 16) ? 0 : 8;

  auto mmToLds = [&](const _Float16* Wt, const float* bs, v16h b0v, v16h b1v) {
#pragma unroll
    for (int ot = 0; ot < 4; ++ot) {
      v8f acc = {};
      acc = wmma16(ldsA(Wt, 64, ot * 16, 0, lane), b0v, acc);
      acc = wmma16(ldsA(Wt, 64, ot * 16, 32, lane), b1v, acc);
#pragma unroll
      for (int r = 0; r < 8; ++r) {
        int o = ot * 16 + r + rsel;
        t1[n * 64 + o] = (_Float16)fast_tanh(acc[r] + bs[o]);  // [pos][ch]
      }
    }
  };

  // Z branch
  mmToLds(smW[0], smB[0], xb0, xb1);
  v16h tb0 = ldBt(t1, 0, 64, lane);
  v16h tb1 = ldBt(t1, 32, 64, lane);
  v8f zacc[4];
#pragma unroll
  for (int ot = 0; ot < 4; ++ot) {
    v8f acc = {};
    acc = wmma16(ldsA(smW[1], 64, ot * 16, 0, lane), tb0, acc);
    acc = wmma16(ldsA(smW[1], 64, ot * 16, 32, lane), tb1, acc);
    zacc[ot] = acc;
  }
  // F branch (reuses x fragments + t1 scratch)
  mmToLds(smW[2], smB[2], xb0, xb1);
  tb0 = ldBt(t1, 0, 64, lane);
  tb1 = ldBt(t1, 32, 64, lane);
  _Float16* op = xg + (long)b * Cc * HW + pi;
#pragma unroll
  for (int ot = 0; ot < 4; ++ot) {
    v8f acc = {};
    acc = wmma16(ldsA(smW[3], 64, ot * 16, 0, lane), tb0, acc);
    acc = wmma16(ldsA(smW[3], 64, ot * 16, 32, lane), tb1, acc);
#pragma unroll
    for (int r = 0; r < 8; ++r) {
      int o = ot * 16 + r + rsel;
      float zf = fast_tanh(zacc[ot][r] + smB[1][o]);
      float ff = sigmoidf(acc[r] + smB[3][o]);
      op[(long)o * HW + n] = (_Float16)((1.0f - ff) * zf);
    }
  }
}

// ---- 3x3 partial conv on first 16 channels (small; VALU) -------------------

__global__ void __launch_bounds__(256) k_pconv(const _Float16* __restrict__ xg,
                                               const float* __restrict__ pw,
                                               _Float16* __restrict__ xc) {
  __shared__ float sw[DC * DC * 9];
  for (int i = threadIdx.x; i < DC * DC * 9; i += 256) sw[i] = pw[i];
  __syncthreads();
  long i = (long)blockIdx.x * 256 + threadIdx.x;  // 16*B*HW outputs
  int w = (int)(i & (Ww - 1));
  int h = (int)((i >> 7) & (Hh - 1));
  int o = (int)((i >> 14) & (DC - 1));
  int b = (int)(i >> 18);
  float acc = 0.0f;
  for (int c = 0; c < DC; ++c) {
    const _Float16* xpc = xg + (long)(b * Cc + c) * HW;
    const float* wc = &sw[(o * DC + c) * 9];
#pragma unroll
    for (int ky = 0; ky < 3; ++ky) {
      int gh = h + ky - 1;
      if (gh < 0 || gh >= Hh) continue;
#pragma unroll
      for (int kx = 0; kx < 3; ++kx) {
        int gw = w + kx - 1;
        if (gw < 0 || gw >= Ww) continue;
        acc += wc[ky * 3 + kx] * (float)xpc[gh * Ww + gw];
      }
    }
  }
  xc[((long)(b * DC + o) * Hh + h) * Ww + w] = (_Float16)acc;
}

// ---- Fused FFN: lin1+GELU -> depthwise 3x3+GELU -> *h2 -> lin2 + residual.
// One 16x16 spatial tile per block; h kept on an 18x18 halo region in LDS
// (~312KB of the 320KB/WGP LDS used) so h/g never touch HBM. lin2 weights
// are staged into LDS by the Tensor Data Mover when available. --------------

__global__ void __launch_bounds__(256) k_ffn(
    const _Float16* __restrict__ xg, const _Float16* __restrict__ xc,
    const float* __restrict__ l1w, const float* __restrict__ l1b,
    const float* __restrict__ dww, const float* __restrict__ dwb,
    const float* __restrict__ l2w, const float* __restrict__ l2b,
    _Float16* __restrict__ xout) {
  extern __shared__ char smem[];
  _Float16* smW1 = (_Float16*)smem;         // [256][64]  A-layout lin1
  _Float16* smW2 = smW1 + 256 * 64;         // [64][128]  A-layout lin2
  _Float16* smH  = smW2 + 64 * 128;         // [256][324] h on halo region
  _Float16* smG  = smH + 256 * RR;          // [256 pos][128 ch] gated g (transposed)
  float* smDW  = (float*)(smG + 256 * 128); // [128][9]
  float* smB1  = smDW + 128 * 9;            // [256]
  float* smDB  = smB1 + 256;                // [128]
  float* smB2  = smDB + 128;                // [64]
  float* smRaw = smB2 + 64;                 // [128*64] TDM staging for lin2_w

  int tid = threadIdx.x;
  for (int i = tid; i < 256 * 64; i += 256) {
    int o = i >> 6, c = i & 63;
    smW1[i] = (_Float16)l1w[c * 256 + o];   // einsum 'bchw,ck' -> A(m=k_out, k=c)
  }
  for (int i = tid; i < 128 * 9; i += 256) smDW[i] = dww[i];
  smB1[tid] = l1b[tid];
  if (tid < 128) smDB[tid] = dwb[tid];
  if (tid < 64)  smB2[tid] = l2b[tid];

#if HAVE_TDM
  // Stage lin2_w (128x64 fp32, 32KB) into LDS via the Tensor Data Mover.
  // D# group0: count=1 | lds_addr | global_addr[56:0] | type=2 ("image").
  // D# group1: data_size=4B, tensor_dim0=64, tensor_dim1=128, tile=64x128,
  //            dim0_stride=64 (whole matrix as one 2D tile).
  if (tid == 0) {
    unsigned long long ga = (unsigned long long)(uintptr_t)l2w;
    unsigned lds_off = (unsigned)(uintptr_t)(void*)smRaw;  // low 32b = LDS byte offset
    u32x4 g0 = {1u, lds_off, (unsigned)(ga & 0xffffffffu),
                (unsigned)((ga >> 32) & 0x1ffffffu) | 0x80000000u};
    i32x8 g1 = {0x00020000,          // data_size=4B, no multicast/pad/iterate
                (int)(64u << 16),    // tensor_dim0[15:0] in bits [63:48]... low half
                (int)(128u << 16),   // tensor_dim1 (bits [111:80] low part)
                (int)(64u << 16),    // tile_dim0 at [127:112]
                128,                 // tile_dim1 at [143:128]
                64,                  // tensor_dim0_stride at [207:160]
                0, 0};
    i32x4 gz = {0, 0, 0, 0};
#if defined(__clang_major__) && __clang_major__ >= 23
    i32x8 gz8 = {0, 0, 0, 0, 0, 0, 0, 0};
    __builtin_amdgcn_tensor_load_to_lds(g0, g1, gz, gz, gz8, 0);
#else
    __builtin_amdgcn_tensor_load_to_lds(g0, g1, gz, gz, 0);
#endif
    __builtin_amdgcn_s_wait_tensorcnt(0);
  }
  __syncthreads();
  for (int i = tid; i < 64 * 128; i += 256) {
    int c = i >> 7, k = i & 127;
    smW2[i] = (_Float16)smRaw[k * 64 + c];  // A(m=c_out, k)
  }
#else
  for (int i = tid; i < 64 * 128; i += 256) {
    int c = i >> 7, k = i & 127;
    smW2[i] = (_Float16)l2w[k * 64 + c];
  }
#endif
  __syncthreads();

  int bid = blockIdx.x;
  int b = bid >> 6;
  int t = bid & 63;
  int h0 = (t >> 3) * 16;
  int w0 = (t & 7) * 16;

  int lane = tid & 31, wv = tid >> 5;
  int n = lane & 15;
  int rsel = (lane < 16) ? 0 : 8;

  // Stage 1: h = gelu(lin1(concat(xc, xg[16:]))) on 18x18 region (21 pos-tiles)
  for (int pt = wv; pt < 21; pt += 8) {
    v16h b0{}, b1{};
    int rp = pt * 16 + n;
    {
      int rh = rp / RG, rw = rp % RG;
      int gh = h0 + rh - 1, gw = w0 + rw - 1;
      bool ok = (rp < RR) && gh >= 0 && gh < Hh && gw >= 0 && gw < Ww;
      int k0 = (lane < 16) ? 0 : 16;
      if (ok) {
        long pix = (long)gh * Ww + gw;
#pragma unroll
        for (int e = 0; e < 16; ++e) {
          int c = k0 + e;  // 0..31
          b0[e] = (c < DC) ? xc[(long)(b * DC + c) * HW + pix]
                           : xg[(long)(b * Cc + c) * HW + pix];
          b1[e] = xg[(long)(b * Cc + 32 + k0 + e) * HW + pix];
        }
      }
    }
    for (int ot = 0; ot < 16; ++ot) {
      v8f acc = {};
      acc = wmma16(ldsA(smW1, 64, ot * 16, 0, lane), b0, acc);
      acc = wmma16(ldsA(smW1, 64, ot * 16, 32, lane), b1, acc);
      if (rp < RR) {
#pragma unroll
        for (int r = 0; r < 8; ++r) {
          int o = ot * 16 + r + rsel;
          smH[o * RR + rp] = (_Float16)geluf(acc[r] + smB1[o]);
        }
      }
    }
  }
  __syncthreads();

  // Stage 2: g = gelu(dwconv3x3(h1)+b) * h2 (all operands in LDS);
  // g stored TRANSPOSED [pos][ch] so stage-3 B loads are contiguous b128.
  for (int i = tid; i < HID * 256; i += 256) {
    int c = i >> 8;
    int p = i & 255;
    int py = p >> 4, px = p & 15;
    const float* wc = &smDW[c * 9];
    float acc = 0.0f;
#pragma unroll
    for (int dy = 0; dy < 3; ++dy) {
      int gh = h0 + py + dy - 1;
      if (gh < 0 || gh >= Hh) continue;
#pragma unroll
      for (int dx = 0; dx < 3; ++dx) {
        int gw = w0 + px + dx - 1;
        if (gw < 0 || gw >= Ww) continue;
        acc += wc[dy * 3 + dx] * (float)smH[c * RR + (py + dy) * RG + (px + dx)];
      }
    }
    float h1 = geluf(acc + smDB[c]);
    float h2 = (float)smH[(HID + c) * RR + (py + 1) * RG + (px + 1)];
    smG[p * HID + c] = (_Float16)(h1 * h2);
  }
  __syncthreads();

  // Stage 3: x_out = lin2(g) + bias + residual (residual = pre-conv xg)
  for (int pt = wv; pt < 16; pt += 8) {
    v16h gb[4];
#pragma unroll
    for (int kt = 0; kt < 4; ++kt) gb[kt] = ldBt(smG + (pt * 16) * HID, kt * 32, HID, lane);
    int p = pt * 16 + n;
    int py = p >> 4, px = p & 15;
    int gh = h0 + py, gw = w0 + px;
#pragma unroll
    for (int ot = 0; ot < 4; ++ot) {
      v8f acc = {};
#pragma unroll
      for (int kt = 0; kt < 4; ++kt)
        acc = wmma16(ldsA(smW2, 128, ot * 16, kt * 32, lane), gb[kt], acc);
#pragma unroll
      for (int r = 0; r < 8; ++r) {
        int c = ot * 16 + r + rsel;
        long idx = ((long)(b * Cc + c) * Hh + gh) * Ww + gw;
        float res = (float)xg[idx];
        xout[idx] = (_Float16)(acc[r] + smB2[c] + res);
      }
    }
  }
}

// ---- Final 3x3 conv (64 -> 1) on (x + x_init) ------------------------------

__global__ void __launch_bounds__(256) k_out(const _Float16* __restrict__ xf,
                                             const float* __restrict__ xinit,
                                             const float* __restrict__ ow,
                                             const float* __restrict__ ob,
                                             float* __restrict__ out) {
  __shared__ float sw[Cc * 9];
  for (int i = threadIdx.x; i < Cc * 9; i += 256) sw[i] = ow[i];
  __syncthreads();
  long i = (long)blockIdx.x * 256 + threadIdx.x;  // B*HW outputs
  int w = (int)(i & (Ww - 1));
  int h = (int)((i >> 7) & (Hh - 1));
  int b = (int)(i >> 14);
  float acc = ob[0];
  for (int c = 0; c < Cc; ++c) {
    long basei = (long)(b * Cc + c) * HW;
#pragma unroll
    for (int ky = 0; ky < 3; ++ky) {
      int gh = h + ky - 1;
      if (gh < 0 || gh >= Hh) continue;
#pragma unroll
      for (int kx = 0; kx < 3; ++kx) {
        int gw = w + kx - 1;
        if (gw < 0 || gw >= Ww) continue;
        long off = basei + gh * Ww + gw;
        acc += sw[c * 9 + ky * 3 + kx] * ((float)xf[off] + xinit[off]);
      }
    }
  }
  out[i] = acc;
}

// ---- host ------------------------------------------------------------------

extern "C" void kernel_launch(void* const* d_in, const int* in_sizes, int n_in,
                              void* d_out, int out_size, void* d_ws, size_t ws_size,
                              hipStream_t stream) {
  (void)in_sizes; (void)n_in; (void)out_size; (void)ws_size;
  const float* x   = (const float*)d_in[0];
  const float* wz1 = (const float*)d_in[1];
  const float* bz1 = (const float*)d_in[2];
  const float* wz2 = (const float*)d_in[3];
  const float* bz2 = (const float*)d_in[4];
  const float* wf1 = (const float*)d_in[5];
  const float* bf1 = (const float*)d_in[6];
  const float* wf2 = (const float*)d_in[7];
  const float* bf2 = (const float*)d_in[8];
  const float* pcw = (const float*)d_in[9];
  const float* l1w = (const float*)d_in[10];
  const float* l1b = (const float*)d_in[11];
  const float* dww = (const float*)d_in[12];
  const float* dwb = (const float*)d_in[13];
  const float* l2w = (const float*)d_in[14];
  const float* l2b = (const float*)d_in[15];
  const float* ow  = (const float*)d_in[16];
  const float* ob  = (const float*)d_in[17];

  _Float16* buf0 = (_Float16*)d_ws;                 // x (layer input/output), f16
  _Float16* buf1 = buf0 + (long)Bn * Cc * HW;       // gated x / residual, f16
  _Float16* buf2 = buf1 + (long)Bn * Cc * HW;       // pconv output (16 ch), f16

  k_cvt<<<8192, 256, 0, stream>>>(x, buf0, (long)Bn * Cc * HW);

  constexpr size_t FFN_SMEM =
      (size_t)(256 * 64 + 64 * 128 + 256 * RR + 256 * 128) * sizeof(_Float16) +
      (size_t)(128 * 9 + 256 + 128 + 64 + 128 * 64) * sizeof(float);  // ~312 KB

  for (int l = 0; l < 3; ++l) {
    k_gate<<<2048, 256, 0, stream>>>(buf0,
        wz1 + l * 4096, bz1 + l * 64, wz2 + l * 4096, bz2 + l * 64,
        wf1 + l * 4096, bf1 + l * 64, wf2 + l * 4096, bf2 + l * 64, buf1);
    k_pconv<<<(Bn * DC * HW) / 256, 256, 0, stream>>>(buf1, pcw + l * DC * DC * 9, buf2);
    k_ffn<<<Bn * 64, 256, FFN_SMEM, stream>>>(buf1, buf2,
        l1w + l * Cc * 2 * HID, l1b + l * 2 * HID,
        dww + l * HID * 9, dwb + l * HID,
        l2w + l * HID * Cc, l2b + l * Cc, buf0);
  }
  k_out<<<(Bn * HW) / 256, 256, 0, stream>>>(buf0, x, ow, ob, (float*)d_out);
}
